// Block_37237366456420
// MI455X (gfx1250) — matrix operand order repaired and checked by reading, not compiled, not run on hardware
//
#include <hip/hip_runtime.h>
#include <cstdint>
#include <cstddef>

// ---------------- CDNA5 WMMA types ----------------
typedef __attribute__((ext_vector_type(16))) __bf16 v16bf;
typedef __attribute__((ext_vector_type(8)))  float  v8f;
typedef __attribute__((ext_vector_type(4)))  unsigned int uvec4;

struct Frag { uvec4 lo, hi; };   // 32 bytes = 16 bf16 = one A/B fragment

static __device__ __forceinline__ v8f wmma_bf16(const Frag& a, const Frag& b, v8f c) {
  return __builtin_amdgcn_wmma_f32_16x16x32_bf16(
      /*neg_a=*/false, __builtin_bit_cast(v16bf, a),
      /*neg_b=*/false, __builtin_bit_cast(v16bf, b),
      /*c_mod=*/(short)0, c, /*reuse_a=*/false, /*reuse_b=*/false);
}

static __device__ __forceinline__ unsigned short f2bf(float f) {
  unsigned int u = __float_as_uint(f);
  u += 0x7fffu + ((u >> 16) & 1u);          // round-to-nearest-even
  return (unsigned short)(u >> 16);
}

// CDNA5 async memory->LDS copy (ASYNCcnt-tracked), 16 bytes per lane.
static __device__ __forceinline__ void async_ld16(const unsigned short* g, unsigned lds) {
  asm volatile("global_load_async_to_lds_b128 %0, %1, off"
               :: "v"(lds), "v"((unsigned long long)(uintptr_t)g)
               : "memory");
}
static __device__ __forceinline__ void wait_async() {
  asm volatile("s_wait_asynccnt 0" ::: "memory");
}
// 32-bit LDS offset of a shared-memory pointer (generic ptr low 32 bits).
static __device__ __forceinline__ unsigned lds_off(const void* p) {
  return (unsigned)(uintptr_t)p;
}

#define BDIM  1024
#define SEQ   2048
#define BATCH 2
#define NHEAD 16
#define HD    64
#define FFDIM 4096
#define ROWS  (BATCH * SEQ)   // 4096

// ---------------- fp32 -> bf16 convert ----------------
__global__ void cvt_bf16_kernel(const float* __restrict__ in,
                                unsigned short* __restrict__ out, int n) {
  int i = blockIdx.x * 256 + threadIdx.x;
  if (i < n) out[i] = f2bf(in[i]);
}

// ---------------- LayerNorm (fp32 in -> bf16 out) ----------------
__global__ __launch_bounds__(256) void ln_kernel(const float* __restrict__ x,
                                                 const float* __restrict__ g,
                                                 const float* __restrict__ be,
                                                 unsigned short* __restrict__ out) {
  __shared__ float rs[256], rq[256];
  const int row = blockIdx.x, tid = threadIdx.x;
  const float* xr = x + (size_t)row * BDIM;
  float v[4], s = 0.f, q = 0.f;
#pragma unroll
  for (int i = 0; i < 4; i++) { v[i] = xr[tid + i * 256]; s += v[i]; q += v[i] * v[i]; }
  rs[tid] = s; rq[tid] = q;
  __syncthreads();
  for (int st = 128; st > 0; st >>= 1) {
    if (tid < st) { rs[tid] += rs[tid + st]; rq[tid] += rq[tid + st]; }
    __syncthreads();
  }
  const float mu   = rs[0] * (1.f / BDIM);
  const float var  = rq[0] * (1.f / BDIM) - mu * mu;
  const float rstd = rsqrtf(var + 1e-5f);
#pragma unroll
  for (int i = 0; i < 4; i++) {
    int c = tid + i * 256;
    out[(size_t)row * BDIM + c] = f2bf((v[i] - mu) * rstd * g[c] + be[c]);
  }
}

// ---------------- Generic bf16 GEMM: C = A[MxK] @ B[KxN] ----------------
// mode 0: store bf16               (QKV)
// mode 1: +bias +resid, store f32  (proj, fc2)
// mode 2: +bias, GELU, store bf16  (fc1)
// Double-buffered LDS tiles filled with GLOBAL_LOAD_ASYNC_TO_LDS_B128.
__global__ __launch_bounds__(256) void gemm_bf16_kernel(
    const unsigned short* __restrict__ A, const unsigned short* __restrict__ Bw,
    float* __restrict__ outF, unsigned short* __restrict__ outB,
    const float* __restrict__ bias, const float* __restrict__ resid,
    int M, int N, int K, int mode) {
  __shared__ __attribute__((aligned(16))) unsigned short As[2][128 * 40];  // 128 x 32 k, stride 40
  __shared__ __attribute__((aligned(16))) unsigned short Bs[2][32 * 136];  // 32 k x 128, stride 136
  const int tid = threadIdx.x;
  const int lane = tid & 31, wave = tid >> 5;
  const int hi = lane >> 4, l16 = lane & 15;
  const int wr = wave >> 2, wc = wave & 3;           // 2x4 wave grid
  const int m0 = blockIdx.y * 128, n0 = blockIdx.x * 128;

  const v8f vz = {0.f, 0.f, 0.f, 0.f, 0.f, 0.f, 0.f, 0.f};
  v8f acc[4][2];
#pragma unroll
  for (int s = 0; s < 4; s++)
#pragma unroll
    for (int t = 0; t < 2; t++) acc[s][t] = vz;

  // per-thread copy assignments: 2 x 16B chunks of A-tile, 2 of B-tile
  const int ca0 = tid, ca1 = tid + 256;
  const int ar0 = ca0 >> 2, ao0 = (ca0 & 3) * 8;
  const int ar1 = ca1 >> 2, ao1 = (ca1 & 3) * 8;
  const int br0 = ca0 >> 4, bo0 = (ca0 & 15) * 8;
  const int br1 = ca1 >> 4, bo1 = (ca1 & 15) * 8;
  const unsigned short* ag0 = A + (size_t)(m0 + ar0) * K + ao0;
  const unsigned short* ag1 = A + (size_t)(m0 + ar1) * K + ao1;
  const unsigned short* bg0 = Bw + (size_t)br0 * N + n0 + bo0;
  const unsigned short* bg1 = Bw + (size_t)br1 * N + n0 + bo1;
  const unsigned aL0 = lds_off(&As[0][0] + ar0 * 40 + ao0);
  const unsigned aL1 = lds_off(&As[0][0] + ar1 * 40 + ao1);
  const unsigned bL0 = lds_off(&Bs[0][0] + br0 * 136 + bo0);
  const unsigned bL1 = lds_off(&Bs[0][0] + br1 * 136 + bo1);
  const unsigned aBuf = 128 * 40 * 2;   // bytes per A buffer
  const unsigned bBuf = 32 * 136 * 2;   // bytes per B buffer
  const size_t bstep = (size_t)32 * N;

  // prologue: fill buffer 0
  async_ld16(ag0, aL0); async_ld16(ag1, aL1);
  async_ld16(bg0, bL0); async_ld16(bg1, bL1);
  ag0 += 32; ag1 += 32; bg0 += bstep; bg1 += bstep;

  const int iters = K >> 5;
  for (int it = 0; it < iters; ++it) {
    const int buf = it & 1;
    wait_async();
    __syncthreads();
    if (it + 1 < iters) {                 // prefetch next tile into other buffer
      const unsigned nb = (unsigned)(buf ^ 1);
      async_ld16(ag0, aL0 + nb * aBuf); async_ld16(ag1, aL1 + nb * aBuf);
      async_ld16(bg0, bL0 + nb * bBuf); async_ld16(bg1, bL1 + nb * bBuf);
      ag0 += 32; ag1 += 32; bg0 += bstep; bg1 += bstep;
    }

    const unsigned short* Ab = As[buf];
    const unsigned short* Bb = Bs[buf];
    Frag af[4], bf[2];
#pragma unroll
    for (int s = 0; s < 4; s++) {
      const unsigned short* p = Ab + (wr * 64 + s * 16 + l16) * 40;
      af[s].lo = *(const uvec4*)(p + hi * 8);
      af[s].hi = *(const uvec4*)(p + 16 + hi * 8);
    }
#pragma unroll
    for (int t = 0; t < 2; t++) {
      const unsigned short* p = Bb + wc * 32 + t * 16 + l16;
      unsigned short e[16];
#pragma unroll
      for (int i = 0; i < 16; i++) e[i] = p[(i < 8 ? hi * 8 + i : 16 + hi * 8 + (i - 8)) * 136];
      bf[t] = __builtin_bit_cast(Frag, e);
    }
#pragma unroll
    for (int s = 0; s < 4; s++)
#pragma unroll
      for (int t = 0; t < 2; t++) acc[s][t] = wmma_bf16(af[s], bf[t], acc[s][t]);
  }

  // epilogue (C layout: vgpr r -> row base + r + 8*hi, col = lane&15)
#pragma unroll
  for (int s = 0; s < 4; s++) {
#pragma unroll
    for (int t = 0; t < 2; t++) {
      const int rowb = m0 + wr * 64 + s * 16 + 8 * hi;
      const int col  = n0 + wc * 32 + t * 16 + l16;
#pragma unroll
      for (int r = 0; r < 8; r++) {
        float v = acc[s][t][r];
        size_t off = (size_t)(rowb + r) * N + col;
        if (mode == 0) {
          outB[off] = f2bf(v);
        } else if (mode == 1) {
          outF[off] = v + bias[col] + resid[off];
        } else {
          v += bias[col];
          float ge = 0.5f * v * (1.0f + erff(v * 0.70710678118654752f));
          outB[off] = f2bf(ge);
        }
      }
    }
  }
}

// ---------------- QKV split: qkv[B,N,3,H,d] -> Q[bh,n,d], Kt[bh,d,n], V[bh,n,d] ----------------
__global__ void qkv_split_kernel(const unsigned short* __restrict__ qkv,
                                 unsigned short* __restrict__ Qd,
                                 unsigned short* __restrict__ Ktd,
                                 unsigned short* __restrict__ Vd) {
  size_t idx = (size_t)blockIdx.x * 256 + threadIdx.x;
  int row = (int)(idx / 3072);
  int c   = (int)(idx % 3072);
  int m = c >> 10, rem = c & 1023;
  int h = rem >> 6, dd = rem & 63;
  int b = row >> 11, n = row & 2047;
  int bh = b * NHEAD + h;
  unsigned short val = qkv[idx];
  if (m == 0)       Qd [((size_t)bh * SEQ + n) * HD + dd] = val;
  else if (m == 1)  Ktd[((size_t)bh * HD + dd) * SEQ + n] = val;
  else              Vd [((size_t)bh * SEQ + n) * HD + dd] = val;
}

// ---------------- Flash attention (bf16 WMMA, f32 online softmax) ----------------
// grid: (SEQ/128, B*H). 8 waves; each wave owns 16 Q rows, streams K/V in 32-key
// chunks staged via async-to-LDS double buffering.
__global__ __launch_bounds__(256) void flash_attn_kernel(
    const unsigned short* __restrict__ Q, const unsigned short* __restrict__ Kt,
    const unsigned short* __restrict__ V, unsigned short* __restrict__ O) {
  __shared__ __attribute__((aligned(16))) unsigned short Ks[2][64 * 40];    // Kt tile: 64 d x 32 n
  __shared__ __attribute__((aligned(16))) unsigned short Vs[2][32 * 72];    // V tile : 32 n x 64 d
  __shared__ __attribute__((aligned(16))) unsigned short Ps[8][16 * 40];    // per-wave P 16x32

  const int tid = threadIdx.x, lane = tid & 31, wave = tid >> 5;
  const int hi = lane >> 4, l16 = lane & 15;
  const int bh = blockIdx.y;
  const int n0 = blockIdx.x * 128 + wave * 16;       // wave's Q-row base
  const float scale = 0.125f;                        // 1/sqrt(64)

  Frag qa[2];                                        // Q rows n0+l16, d=64 as 2 A-frags
  {
    const unsigned short* qb = Q + ((size_t)bh * SEQ + n0 + l16) * HD;
#pragma unroll
    for (int dc = 0; dc < 2; dc++) {
      qa[dc].lo = *(const uvec4*)(qb + dc * 32 + hi * 8);
      qa[dc].hi = *(const uvec4*)(qb + dc * 32 + 16 + hi * 8);
    }
  }

  const v8f vz = {0.f, 0.f, 0.f, 0.f, 0.f, 0.f, 0.f, 0.f};
  v8f oacc[4];
#pragma unroll
  for (int t = 0; t < 4; t++) oacc[t] = vz;
  float m8[8], l8[8];
#pragma unroll
  for (int r = 0; r < 8; r++) { m8[r] = -1e30f; l8[r] = 0.f; }

  // per-thread async-copy assignments (one 16B chunk of each tile)
  const int kr = tid >> 2, kc = (tid & 3) * 8;       // Kt tile 64x32
  const int vr = tid >> 3, vc = (tid & 7) * 8;       // V  tile 32x64
  const unsigned short* kg = Kt + ((size_t)bh * HD + kr) * SEQ + kc;
  const unsigned short* vg = V + ((size_t)bh * SEQ + vr) * HD + vc;
  const unsigned kL = lds_off(&Ks[0][0] + kr * 40 + kc);
  const unsigned vL = lds_off(&Vs[0][0] + vr * 72 + vc);
  const unsigned kBuf = 64 * 40 * 2, vBuf = 32 * 72 * 2;

  async_ld16(kg, kL); async_ld16(vg, vL);
  kg += 32; vg += 32 * HD;

  const int iters = SEQ / 32;
  for (int it = 0; it < iters; ++it) {
    const int buf = it & 1;
    wait_async();
    __syncthreads();
    if (it + 1 < iters) {
      const unsigned nb = (unsigned)(buf ^ 1);
      async_ld16(kg, kL + nb * kBuf);
      async_ld16(vg, vL + nb * vBuf);
      kg += 32; vg += 32 * HD;
    }

    // S = scale * Q Kt : one 16x32 strip = 2 col-tiles x 2 d-chunks
    v8f s[2];
#pragma unroll
    for (int j = 0; j < 2; j++) {
      v8f a = vz;
#pragma unroll
      for (int dc = 0; dc < 2; dc++) {
        Frag kb;
        const unsigned short* p = Ks[buf] + dc * 32 * 40 + j * 16 + l16;
        unsigned short e[16];
#pragma unroll
        for (int i = 0; i < 16; i++) e[i] = p[(i < 8 ? hi * 8 + i : 16 + hi * 8 + (i - 8)) * 40];
        kb = __builtin_bit_cast(Frag, e);
        a = wmma_bf16(qa[dc], kb, a);
      }
#pragma unroll
      for (int r = 0; r < 8; r++) a[r] *= scale;
      s[j] = a;
    }

    // online softmax per row (rows split across half-waves; shfl masks <16 stay in-half)
    float pnew[2][8];
#pragma unroll
    for (int r = 0; r < 8; r++) {
      float mx = fmaxf(s[0][r], s[1][r]);
#pragma unroll
      for (int k = 1; k < 16; k <<= 1) mx = fmaxf(mx, __shfl_xor(mx, k, 32));
      float mn   = fmaxf(m8[r], mx);
      float corr = __expf(m8[r] - mn);
      float p0 = __expf(s[0][r] - mn);
      float p1 = __expf(s[1][r] - mn);
      float rs = p0 + p1;
#pragma unroll
      for (int k = 1; k < 16; k <<= 1) rs += __shfl_xor(rs, k, 32);
      l8[r] = l8[r] * corr + rs;
      m8[r] = mn;
      pnew[0][r] = p0; pnew[1][r] = p1;
#pragma unroll
      for (int t = 0; t < 4; t++) oacc[t][r] *= corr;
    }

    // P: C-layout -> LDS -> A-fragment
#pragma unroll
    for (int r = 0; r < 8; r++) {
      unsigned short* pr = Ps[wave] + (r + 8 * hi) * 40;
      pr[l16]      = f2bf(pnew[0][r]);
      pr[16 + l16] = f2bf(pnew[1][r]);
    }
    __syncthreads();

    Frag pa;
    {
      const unsigned short* p = Ps[wave] + l16 * 40;
      pa.lo = *(const uvec4*)(p + hi * 8);
      pa.hi = *(const uvec4*)(p + 16 + hi * 8);
    }
#pragma unroll
    for (int t = 0; t < 4; t++) {       // O += P @ V (4 col-tiles over d=64)
      Frag vb;
      const unsigned short* p = Vs[buf] + t * 16 + l16;
      unsigned short e[16];
#pragma unroll
      for (int i = 0; i < 16; i++) e[i] = p[(i < 8 ? hi * 8 + i : 16 + hi * 8 + (i - 8)) * 72];
      vb = __builtin_bit_cast(Frag, e);
      oacc[t] = wmma_bf16(pa, vb, oacc[t]);
    }
  }

  const int b = bh >> 4, h = bh & 15;
#pragma unroll
  for (int t = 0; t < 4; t++) {
#pragma unroll
    for (int r = 0; r < 8; r++) {
      int n = n0 + r + 8 * hi;
      size_t off = ((size_t)b * SEQ + n) * BDIM + h * HD + t * 16 + l16;
      O[off] = f2bf(oacc[t][r] / l8[r]);
    }
  }
}

// ---------------- launcher ----------------
extern "C" void kernel_launch(void* const* d_in, const int* in_sizes, int n_in,
                              void* d_out, int out_size, void* d_ws, size_t ws_size,
                              hipStream_t stream) {
  (void)in_sizes; (void)n_in; (void)out_size; (void)ws_size;
  const float* x      = (const float*)d_in[0];
  const float* ln1_g  = (const float*)d_in[1];
  const float* ln1_b  = (const float*)d_in[2];
  const float* qkv_w  = (const float*)d_in[3];
  const float* proj_w = (const float*)d_in[4];
  const float* proj_b = (const float*)d_in[5];
  const float* ln2_g  = (const float*)d_in[6];
  const float* ln2_b  = (const float*)d_in[7];
  const float* fc1_w  = (const float*)d_in[8];
  const float* fc1_b  = (const float*)d_in[9];
  const float* fc2_w  = (const float*)d_in[10];
  const float* fc2_b  = (const float*)d_in[11];
  float* out = (float*)d_out;

  char* ws = (char*)d_ws;   // peak usage ~92.3 MB with buffer reuse
  unsigned short* w_qkv  = (unsigned short*)(ws + 0);          // 6 MB
  unsigned short* w_proj = (unsigned short*)(ws + 6291456);    // 2 MB
  unsigned short* w_fc1  = (unsigned short*)(ws + 8388608);    // 8 MB
  unsigned short* w_fc2  = (unsigned short*)(ws + 16777216);   // 8 MB
  unsigned short* h_bf   = (unsigned short*)(ws + 25165824);   // 8 MB (LN1 out, reused by LN2)
  unsigned short* qkv_bf = (unsigned short*)(ws + 33554432);   // 24 MB
  unsigned short* o_bf   = qkv_bf;                             // reuse after split
  unsigned short* q_bf   = (unsigned short*)(ws + 58720256);   // 8 MB
  unsigned short* kt_bf  = (unsigned short*)(ws + 67108864);   // 8 MB
  unsigned short* v_bf   = (unsigned short*)(ws + 75497472);   // 8 MB
  unsigned short* ffh_bf = q_bf;                               // 32 MB, reuse q/kt/v region

  cvt_bf16_kernel<<<(1024 * 3072 + 255) / 256, 256, 0, stream>>>(qkv_w, w_qkv, 1024 * 3072);
  cvt_bf16_kernel<<<(1024 * 1024 + 255) / 256, 256, 0, stream>>>(proj_w, w_proj, 1024 * 1024);
  cvt_bf16_kernel<<<(1024 * 4096 + 255) / 256, 256, 0, stream>>>(fc1_w, w_fc1, 1024 * 4096);
  cvt_bf16_kernel<<<(4096 * 1024 + 255) / 256, 256, 0, stream>>>(fc2_w, w_fc2, 4096 * 1024);

  ln_kernel<<<ROWS, 256, 0, stream>>>(x, ln1_g, ln1_b, h_bf);

  gemm_bf16_kernel<<<dim3(24, 32), 256, 0, stream>>>(h_bf, w_qkv, nullptr, qkv_bf,
                                                     nullptr, nullptr, ROWS, 3072, 1024, 0);

  qkv_split_kernel<<<(ROWS * 3072) / 256, 256, 0, stream>>>(qkv_bf, q_bf, kt_bf, v_bf);

  flash_attn_kernel<<<dim3(SEQ / 128, BATCH * NHEAD), 256, 0, stream>>>(q_bf, kt_bf, v_bf, o_bf);

  gemm_bf16_kernel<<<dim3(8, 32), 256, 0, stream>>>(o_bf, w_proj, out, nullptr,
                                                    proj_b, x, ROWS, 1024, 1024, 1);

  ln_kernel<<<ROWS, 256, 0, stream>>>(out, ln2_g, ln2_b, h_bf);

  gemm_bf16_kernel<<<dim3(32, 32), 256, 0, stream>>>(h_bf, w_fc1, nullptr, ffh_bf,
                                                     fc1_b, nullptr, ROWS, FFDIM, 1024, 2);

  gemm_bf16_kernel<<<dim3(8, 32), 256, 0, stream>>>(ffh_bf, w_fc2, out, nullptr,
                                                    fc2_b, out, ROWS, 1024, FFDIM, 1);
}
